// SwarmAttentionHead_11201274708160
// MI455X (gfx1250) — matrix operand order repaired and checked by reading, not compiled, not run on hardware
//
#include <hip/hip_runtime.h>
#include <hip/hip_bf16.h>
#include <stdint.h>

// ---------------------------------------------------------------------------
// SwarmAttentionHead for MI455X (gfx1250): bf16 WMMA flash-attention pipeline
// with async global->LDS double-buffered K/V staging.
//   N = 8192 nodes, D = 256 channels.
// ---------------------------------------------------------------------------

#define NN 8192
#define DD 256

typedef __attribute__((ext_vector_type(16))) __bf16       v16bf;
typedef __attribute__((ext_vector_type(8)))  __bf16       v8bf;
typedef __attribute__((ext_vector_type(4)))  __bf16       v4bf;
typedef __attribute__((ext_vector_type(8)))  float        v8f;
typedef __attribute__((ext_vector_type(4)))  float        f32x4;
typedef __attribute__((ext_vector_type(4)))  unsigned int u32x4;
typedef __attribute__((ext_vector_type(2)))  unsigned int u32x2;

// Async global->LDS path (gfx1250 GLOBAL_LOAD_ASYNC_TO_LDS_B128, ASYNCcnt).
// Probe round 2 showed the builtin exists with signature:
//   (v4i __device__(AS1)*, v4i __shared__(AS3)*, imm offset, imm cpol)
#if defined(__HIP_DEVICE_COMPILE__) && __has_builtin(__builtin_amdgcn_global_load_async_to_lds_b128)
#define ATTN_ASYNC 1
#else
#define ATTN_ASYNC 0
#endif

#if ATTN_ASYNC
typedef int v4i_vs __attribute__((vector_size(16)));
typedef __attribute__((address_space(1))) v4i_vs glob_v4i;
typedef __attribute__((address_space(3))) v4i_vs lds_v4i;
#if __has_builtin(__builtin_amdgcn_s_wait_asynccnt)
#define WAIT_ASYNC() __builtin_amdgcn_s_wait_asynccnt(0)
#else
#define WAIT_ASYNC() asm volatile("s_wait_asynccnt 0" ::: "memory")
#endif
#else
#define WAIT_ASYNC()
#endif

__device__ __forceinline__ void copy16_to_lds(__bf16* l, const __bf16* g) {
#if ATTN_ASYNC
  __builtin_amdgcn_global_load_async_to_lds_b128((glob_v4i*)g, (lds_v4i*)l, 0, 0);
#else
  *(u32x4*)l = *(const u32x4*)g;
#endif
}

// ---------------------------------------------------------------------------
// Fragment loaders (CDNA5 WMMA VGPR layouts, wave32).
// A-matrix 16x32 bf16: lane l<16 -> row l, K in {kb+0..7, kb+16..23};
//                      lane l>=16 -> row l-16, K in {kb+8..15, kb+24..31}.
// B-matrix 32x16 bf16: lane l<16 -> col l, K in {kb+0..7, kb+16..23};
//                      lane l>=16 -> col l-16, K in {kb+8..15, kb+24..31}.
// C/D 16x16 f32: vgpr r: lanes 0-15 = (M=r, N=lane), lanes 16-31 = (M=r+8).
// ---------------------------------------------------------------------------
__device__ __forceinline__ v16bf load_a_frag(const __bf16* base, int rowBase,
                                             int ld, int kb, int lane) {
  const int half = lane >> 4, l16 = lane & 15;
  const __bf16* p = base + (size_t)(rowBase + l16) * ld + kb + half * 8;
  union { v16bf v; u32x4 u[2]; } f;
  f.u[0] = *(const u32x4*)(p);
  f.u[1] = *(const u32x4*)(p + 16);
  return f.v;
}

__device__ __forceinline__ v16bf load_b_frag(const __bf16* base, int colBase,
                                             int ld, int kb, int lane) {
  const int half = lane >> 4, l16 = lane & 15;
  const __bf16* p = base + (size_t)(colBase + l16) * ld + kb + half * 8;
  union { v16bf v; u32x4 u[2]; } f;
  f.u[0] = *(const u32x4*)(p);
  f.u[1] = *(const u32x4*)(p + 16);
  return f.v;
}

__device__ __forceinline__ v8f wmma_bf16(v16bf a, v16bf b, v8f c) {
  // (neg_a, A, neg_b, B, c_mod, C, reuse_a, reuse_b)
  return __builtin_amdgcn_wmma_f32_16x16x32_bf16(false, a, false, b,
                                                 (short)0, c, false, false);
}

// ---------------------------------------------------------------------------
// fp32 -> bf16 conversion (4 elements / thread)
// ---------------------------------------------------------------------------
__global__ void k_cvt_bf16(const float* __restrict__ src, __bf16* __restrict__ dst,
                           int n4) {
  int i = blockIdx.x * blockDim.x + threadIdx.x;
  if (i >= n4) return;
  f32x4 v = ((const f32x4*)src)[i];
  union { v4bf b; u32x2 u; } o;
  o.b[0] = (__bf16)v[0];
  o.b[1] = (__bf16)v[1];
  o.b[2] = (__bf16)v[2];
  o.b[3] = (__bf16)v[3];
  ((u32x2*)dst)[i] = o.u;
}

// ---------------------------------------------------------------------------
// Projection: OUT = A[nRows,256] @ W^T + bias, one 16x16 tile per wave.
//   mode 0: store bf16 [nRows,256], value scaled by `scale`
//   mode 1: store bf16 transposed [256,nRows] (for V)
//   mode 2: store f32  [nRows,256]            (final output)
// ---------------------------------------------------------------------------
__global__ void __launch_bounds__(128, 1)
k_proj(const __bf16* __restrict__ A, const __bf16* __restrict__ W,
       const float* __restrict__ bias, __bf16* __restrict__ outb,
       float* __restrict__ outf, int mode, float scale, int nRows) {
  const int lane = threadIdx.x & 31;
  const int wave = (blockIdx.x * blockDim.x + threadIdx.x) >> 5;
  const int tileD = wave & 15;        // 16 col tiles (D=256)
  const int tileM = wave >> 4;        // row tile
  if (tileM >= nRows / 16) return;
  const int half = lane >> 4, l16 = lane & 15;

  v8f c = {0.f, 0.f, 0.f, 0.f, 0.f, 0.f, 0.f, 0.f};
#pragma unroll
  for (int kb = 0; kb < DD; kb += 32) {
    v16bf a = load_a_frag(A, tileM * 16, DD, kb, lane);
    v16bf b = load_b_frag(W, tileD * 16, DD, kb, lane);  // B(k,n) = W[n][k] -> A@W^T
    c = wmma_bf16(a, b, c);
  }
  const float bv = bias[tileD * 16 + l16];

  if (mode == 0) {
#pragma unroll
    for (int r = 0; r < 8; ++r) {
      float v = (c[r] + bv) * scale;
      outb[(size_t)(tileM * 16 + half * 8 + r) * DD + tileD * 16 + l16] = (__bf16)v;
    }
  } else if (mode == 1) {
    // Transposed store: for a fixed lane, rows r=0..7 are contiguous -> 16B store.
    union { v8bf b; u32x4 u; } pk;
#pragma unroll
    for (int r = 0; r < 8; ++r) pk.b[r] = (__bf16)(c[r] + bv);
    *(u32x4*)(outb + (size_t)(tileD * 16 + l16) * nRows + tileM * 16 + half * 8) = pk.u;
  } else {
#pragma unroll
    for (int r = 0; r < 8; ++r)
      outf[(size_t)(tileM * 16 + half * 8 + r) * DD + tileD * 16 + l16] = c[r] + bv;
  }
}

// ---------------------------------------------------------------------------
// Cooperative staging of one 32-key K/V block into LDS (128 threads).
//   sK[32][256]  <- K[jb..jb+31][0..255]          (16 KB)
//   sV[256][32]  <- Vt[0..255][jb..jb+31]         (16 KB)
// Uses async DMA when available (ASYNCcnt), else register relay.
// ---------------------------------------------------------------------------
__device__ __forceinline__ void stageKV(const __bf16* __restrict__ Kg,
                                        const __bf16* __restrict__ Vtg, int jb,
                                        __bf16* sKbuf, __bf16* sVbuf, int tid) {
#pragma unroll
  for (int it = 0; it < 8; ++it) {
    int chunk = it * 128 + tid;          // 1024 chunks of 8 bf16
    int r = chunk >> 5, c8 = (chunk & 31) * 8;
    copy16_to_lds(sKbuf + r * DD + c8, Kg + (size_t)(jb + r) * DD + c8);
  }
#pragma unroll
  for (int it = 0; it < 8; ++it) {
    int chunk = it * 128 + tid;
    int d = chunk >> 2, c8 = (chunk & 3) * 8;
    copy16_to_lds(sVbuf + d * 32 + c8, Vtg + (size_t)d * NN + jb + c8);
  }
}

// ---------------------------------------------------------------------------
// Flash attention: one wave owns 16 query rows; the 4 waves of a block share
// double-buffered K/V tiles in LDS. Per 32-key block: 16 WMMAs for S = Q@K^T,
// adjacency mask, online softmax, 16 WMMAs for O += P@V.
// ---------------------------------------------------------------------------
__global__ void __launch_bounds__(128, 1)
k_attn(const __bf16* __restrict__ Q, const __bf16* __restrict__ K,
       const __bf16* __restrict__ Vt, const int* __restrict__ adj,
       __bf16* __restrict__ ctx) {
  __shared__ __align__(16) __bf16 sQ[4][16 * DD];   // 8 KB per wave
  __shared__ __align__(16) __bf16 sP[4][16 * 32];   // 1 KB per wave
  __shared__ __align__(16) __bf16 sK[2][32 * DD];   // 16 KB per buffer
  __shared__ __align__(16) __bf16 sV[2][DD * 32];   // 16 KB per buffer

  const int tid  = threadIdx.x;
  const int lane = tid & 31;
  const int warp = tid >> 5;
  const int qBase = (blockIdx.x * 4 + warp) * 16;   // 512 query tiles total
  const int half = lane >> 4, l16 = lane & 15;

  // Stage this wave's 16x256 Q tile into LDS (re-read every key block).
  {
    const __bf16* qg = Q + (size_t)qBase * DD;
#pragma unroll
    for (int i = 0; i < 16; ++i) {
      int idx = (i * 32 + lane) * 8;                // 8 bf16 = 16 B per lane
      *(u32x4*)(&sQ[warp][idx]) = *(const u32x4*)(qg + idx);
    }
  }
  asm volatile("s_wait_dscnt 0" ::: "memory");

  v8f o[16];
  const v8f vzero = {0.f, 0.f, 0.f, 0.f, 0.f, 0.f, 0.f, 0.f};
#pragma unroll
  for (int t = 0; t < 16; ++t) o[t] = vzero;
  float m[8], l[8];
#pragma unroll
  for (int r = 0; r < 8; ++r) { m[r] = -3.0e38f; l[r] = 0.f; }

  const int* adjBase = adj + (size_t)(qBase + half * 8) * NN + l16;

  // Prologue: stage block 0 into buffer 0.
  stageKV(K, Vt, 0, &sK[0][0], &sV[0][0], tid);

  for (int i = 0; i < NN / 32; ++i) {
    const int jb = i * 32;
    const int cur = i & 1;

    WAIT_ASYNC();          // my async loads for sK/sV[cur] have landed
    __syncthreads();       // everyone's landed; everyone done reading [cur^1]

    if (jb + 32 < NN)      // overlap next block's DMA with this block's math
      stageKV(K, Vt, jb + 32, &sK[cur ^ 1][0], &sV[cur ^ 1][0], tid);
    __builtin_prefetch(adjBase + jb + 32, 0, 1);   // next adjacency block

    // ---- S = Q @ K^T over the 32 keys of this block ----
    v8f s0 = vzero, s1 = vzero;
#pragma unroll
    for (int kb = 0; kb < DD; kb += 32) {
      union { v16bf v; u32x4 u[2]; } fa;
      {
        int off = l16 * DD + kb + half * 8;
        fa.u[0] = *(const u32x4*)(&sQ[warp][off]);
        fa.u[1] = *(const u32x4*)(&sQ[warp][off + 16]);
      }
      v16bf bK0 = load_b_frag(&sK[cur][0], 0, DD, kb, lane);
      v16bf bK1 = load_b_frag(&sK[cur][0], 16, DD, kb, lane);
      s0 = wmma_bf16(fa.v, bK0, s0);
      s1 = wmma_bf16(fa.v, bK1, s1);
    }

    // ---- adjacency mask (1/sqrt(D) folded into Q) ----
#pragma unroll
    for (int r = 0; r < 8; ++r) {
      int a0 = adjBase[(size_t)r * NN + jb];
      int a1 = adjBase[(size_t)r * NN + jb + 16];
      if (a0 == 0) s0[r] = -1.0e9f;
      if (a1 == 0) s1[r] = -1.0e9f;
    }

    // ---- online softmax: row stats replicated across each 16-lane half ----
    float alpha[8];
#pragma unroll
    for (int r = 0; r < 8; ++r) {
      float bm = fmaxf(s0[r], s1[r]);
#pragma unroll
      for (int msk = 1; msk < 16; msk <<= 1) bm = fmaxf(bm, __shfl_xor(bm, msk, 16));
      float mn = fmaxf(m[r], bm);
      float al = __expf(m[r] - mn);
      float p0 = __expf(s0[r] - mn);
      float p1 = __expf(s1[r] - mn);
      float rs = p0 + p1;
#pragma unroll
      for (int msk = 1; msk < 16; msk <<= 1) rs += __shfl_xor(rs, msk, 16);
      l[r] = l[r] * al + rs;
      m[r] = mn;
      alpha[r] = al;
      s0[r] = p0;
      s1[r] = p1;
    }
#pragma unroll
    for (int t = 0; t < 16; ++t) {
#pragma unroll
      for (int r = 0; r < 8; ++r) o[t][r] = o[t][r] * alpha[r];
    }

    // ---- route P (C-layout f32) -> A-fragment (bf16) through per-wave LDS ----
#pragma unroll
    for (int r = 0; r < 8; ++r) {
      int row = half * 8 + r;
      sP[warp][row * 32 + l16]      = (__bf16)s0[r];
      sP[warp][row * 32 + 16 + l16] = (__bf16)s1[r];
    }
    asm volatile("s_wait_dscnt 0" ::: "memory");
    union { v16bf v; u32x4 u[2]; } fp;
    {
      int off = l16 * 32 + half * 8;
      fp.u[0] = *(const u32x4*)(&sP[warp][off]);
      fp.u[1] = *(const u32x4*)(&sP[warp][off + 16]);
    }

    // ---- O += P @ V (sV is [d][key] so K is the fast axis) ----
#pragma unroll
    for (int t = 0; t < 16; ++t) {
      v16bf bV = load_b_frag(&sV[cur][0], t * 16, 32, 0, lane);
      o[t] = wmma_bf16(fp.v, bV, o[t]);
    }
    asm volatile("" ::: "memory");  // keep next iter's sP stores behind these loads
  }

  // Normalize and store context (bf16) for the output projection.
#pragma unroll
  for (int r = 0; r < 8; ++r) l[r] = 1.0f / l[r];
#pragma unroll
  for (int t = 0; t < 16; ++t) {
#pragma unroll
    for (int r = 0; r < 8; ++r)
      ctx[(size_t)(qBase + half * 8 + r) * DD + t * 16 + l16] = (__bf16)(o[t][r] * l[r]);
  }
}

// ---------------------------------------------------------------------------
// Host-side orchestration
// ---------------------------------------------------------------------------
extern "C" void kernel_launch(void* const* d_in, const int* in_sizes, int n_in,
                              void* d_out, int out_size, void* d_ws, size_t ws_size,
                              hipStream_t stream) {
  const float* E   = (const float*)d_in[0];
  const int*   adj = (const int*)d_in[1];
  const float* Wq  = (const float*)d_in[2];
  const float* bq  = (const float*)d_in[3];
  const float* Wk  = (const float*)d_in[4];
  const float* bk  = (const float*)d_in[5];
  const float* Wv  = (const float*)d_in[6];
  const float* bv  = (const float*)d_in[7];
  const float* Wo  = (const float*)d_in[8];
  const float* bo  = (const float*)d_in[9];
  float* out = (float*)d_out;

  // Workspace layout (bf16 buffers), ~20.5 MB total.
  char* w = (char*)d_ws;
  __bf16* Ebf = (__bf16*)w;                 w += (size_t)NN * DD * 2;
  __bf16* Wqb = (__bf16*)w;                 w += (size_t)DD * DD * 2;
  __bf16* Wkb = (__bf16*)w;                 w += (size_t)DD * DD * 2;
  __bf16* Wvb = (__bf16*)w;                 w += (size_t)DD * DD * 2;
  __bf16* Wob = (__bf16*)w;                 w += (size_t)DD * DD * 2;
  __bf16* Qs  = (__bf16*)w;                 w += (size_t)NN * DD * 2;
  __bf16* Kb  = (__bf16*)w;                 w += (size_t)NN * DD * 2;
  __bf16* Vt  = (__bf16*)w;                 w += (size_t)NN * DD * 2;  // [D,N]
  __bf16* Ctx = (__bf16*)w;                 w += (size_t)NN * DD * 2;

  // 1) fp32 -> bf16 conversions
  {
    int n4 = NN * DD / 4;
    k_cvt_bf16<<<(n4 + 255) / 256, 256, 0, stream>>>(E, Ebf, n4);
    int w4 = DD * DD / 4;
    k_cvt_bf16<<<(w4 + 255) / 256, 256, 0, stream>>>(Wq, Wqb, w4);
    k_cvt_bf16<<<(w4 + 255) / 256, 256, 0, stream>>>(Wk, Wkb, w4);
    k_cvt_bf16<<<(w4 + 255) / 256, 256, 0, stream>>>(Wv, Wvb, w4);
    k_cvt_bf16<<<(w4 + 255) / 256, 256, 0, stream>>>(Wo, Wob, w4);
  }

  // 2) Q/K/V projections. 8192 tiles -> 8192 waves -> 2048 blocks of 128.
  const int projBlocks = (NN / 16) * (DD / 16) * 32 / 128;
  const float scale = 0.0625f;  // 1/sqrt(256), folded into Q
  k_proj<<<projBlocks, 128, 0, stream>>>(Ebf, Wqb, bq, Qs, nullptr, 0, scale, NN);
  k_proj<<<projBlocks, 128, 0, stream>>>(Ebf, Wkb, bk, Kb, nullptr, 0, 1.0f, NN);
  k_proj<<<projBlocks, 128, 0, stream>>>(Ebf, Wvb, bv, Vt, nullptr, 1, 1.0f, NN);

  // 3) Flash attention: 512 query tiles -> 128 blocks of 4 waves.
  k_attn<<<NN / 16 / 4, 128, 0, stream>>>(Qs, Kb, Vt, adj, Ctx);

  // 4) Output projection, fp32 store to d_out.
  k_proj<<<projBlocks, 128, 0, stream>>>(Ctx, Wob, bo, nullptr, out, 2, 1.0f, NN);
}
